// CrossYearEpisodicMemory_14070312862142
// MI455X (gfx1250) — matrix operand order repaired and checked by reading, not compiled, not run on hardware
//
#include <hip/hip_runtime.h>
#include <hip/hip_bf16.h>
#include <math.h>

// ---------------- problem constants ----------------
#define BB   16
#define TT   96
#define TC   97      // conv output length
#define TPAD 112     // padded to 7*16 columns
#define NN   512
#define DD   128
#define HID  (NN*DD)        // 65536
#define MM   1196
#define MPAD 1200
#define KK   8
#define HH   4
#define KW   12

typedef __attribute__((ext_vector_type(16))) __bf16 v16bf;
typedef __attribute__((ext_vector_type(8)))  float  v8f;
typedef __attribute__((ext_vector_type(2)))  float  v2f;

__device__ inline v8f v8f_zero() {
  v8f r;
#pragma unroll
  for (int i = 0; i < 8; ++i) r[i] = 0.f;
  return r;
}

__device__ inline float gelu_exact(float x) {
  return 0.5f * x * (1.f + erff(x * 0.70710678118654752f));
}

// ---------------- kernel 0: zero the per-batch norm accumulators ----------------
__global__ void init_kernel(float* qnorm2) {
  if (threadIdx.x < BB) qnorm2[threadIdx.x] = 0.f;
}

// ---------------- kernel 1: depthwise conv + GELU -> actT[b][t][n] (bf16, t padded to 112) ----
__global__ __launch_bounds__(128) void conv_kernel(const float* __restrict__ x,
                                                   const float* __restrict__ dw_w,
                                                   const float* __restrict__ dw_b,
                                                   __bf16* __restrict__ actT) {
  const int n = blockIdx.x, b = blockIdx.y, t = threadIdx.x;
  if (t >= TPAD) return;
  float g = 0.f;
  if (t < TC) {
    float acc = dw_b[n];
#pragma unroll
    for (int j = 0; j < KW; ++j) {
      int tt = t - 6 + j;
      if (tt >= 0 && tt < TT)
        acc += x[((size_t)b * TT + tt) * NN + n] * dw_w[n * KW + j];
    }
    g = gelu_exact(acc);
  }
  actT[((size_t)b * TPAD + t) * NN + n] = (__bf16)g;
}

// ---------------- kernel 2: big pointwise GEMM via bf16 WMMA, fused bias+GELU+mean -----
// One wave per block: 16 output rows x 112 cols, K=512.
__global__ __launch_bounds__(32) void gemm_kernel(const float* __restrict__ pw_w,
                                                  const float* __restrict__ pw_b,
                                                  const __bf16* __restrict__ actT,
                                                  float* __restrict__ q_pre) {
  const int lane = threadIdx.x;
  const int lm = lane & 15, half = lane >> 4;
  const int row0 = blockIdx.x * 16;
  const int b = blockIdx.y;

  v8f acc[7];
#pragma unroll
  for (int c = 0; c < 7; ++c) acc[c] = v8f_zero();

  const __bf16* actb = actT + (size_t)b * TPAD * NN;
  const int r = row0 + lm;                       // A row for this lane
  const float* arow = pw_w + (size_t)r * NN;

  for (int k0 = 0; k0 < NN; k0 += 32) {
    __builtin_prefetch(arow + k0 + 32, 0, 3);    // near-cache global_prefetch_b8
    // A tile: lanes 0-15 hold K k0..k0+15, lanes 16-31 hold k0+16..31
    v16bf a;
    const float* p0 = arow + k0 + 16 * half;
    const float* p1 = arow + k0 + 8 + 16 * half;
#pragma unroll
    for (int i = 0; i < 8; ++i) { a[i] = (__bf16)p0[i]; a[i + 8] = (__bf16)p1[i]; }
#pragma unroll
    for (int c = 0; c < 7; ++c) {
      const int t = c * 16 + lm;                 // B column (time index)
      v16bf bm = *reinterpret_cast<const v16bf*>(actb + (size_t)t * NN + k0 + 16 * half);
      acc[c] = __builtin_amdgcn_wmma_f32_16x16x32_bf16(
          false, a, false, bm, (short)0, acc[c], false, false);
    }
  }

  // bias + GELU + mean over t<97, reduce across the 16-lane half
  const int rme = row0 + 8 * half;
#pragma unroll
  for (int g = 0; g < 8; ++g) {
    const float bias = pw_b[rme + g];
    float s = 0.f;
#pragma unroll
    for (int c = 0; c < 7; ++c) {
      const int t = c * 16 + lm;
      const float v = gelu_exact(acc[c][g] + bias);
      s += (t < TC) ? v : 0.f;
    }
    s += __shfl_xor(s, 1); s += __shfl_xor(s, 2);
    s += __shfl_xor(s, 4); s += __shfl_xor(s, 8);
    if (lm == g) q_pre[(size_t)b * HID + rme + g] = s * (1.f / (float)TC);
  }
}

// ---------------- kernel 3: layernorm over D -> q (into d_out), accumulate ||q_b||^2 ----
__global__ __launch_bounds__(128) void ln_kernel(const float* __restrict__ q_pre,
                                                 const float* __restrict__ ln_g,
                                                 const float* __restrict__ ln_b,
                                                 float* __restrict__ qout,
                                                 float* __restrict__ qnorm2) {
  const int n = blockIdx.x, b = blockIdx.y, t = threadIdx.x;
  __shared__ float red[DD];
  const size_t base = ((size_t)b * NN + n) * DD;
  const float v = q_pre[base + t];

  red[t] = v; __syncthreads();
  for (int s = 64; s > 0; s >>= 1) { if (t < s) red[t] += red[t + s]; __syncthreads(); }
  const float mu = red[0] * (1.f / DD); __syncthreads();

  const float d = v - mu;
  red[t] = d * d; __syncthreads();
  for (int s = 64; s > 0; s >>= 1) { if (t < s) red[t] += red[t + s]; __syncthreads(); }
  const float var = red[0] * (1.f / DD); __syncthreads();

  const float y = d * rsqrtf(var + 1e-5f) * ln_g[t] + ln_b[t];
  qout[base + t] = y;

  red[t] = y * y; __syncthreads();
  for (int s = 64; s > 0; s >>= 1) { if (t < s) red[t] += red[t + s]; __syncthreads(); }
  if (t == 0) atomicAdd(&qnorm2[b], red[0]);
}

// ---------------- kernel 4: similarity via f32 WMMA, fused mem-norm + mask + diversity ---
// One block (4 waves) per 16-column memory tile; waves split K=65536.
__global__ __launch_bounds__(128) void sim_kernel(const float* __restrict__ qv,
                                                  const float* __restrict__ mem,
                                                  const float* __restrict__ qnorm2,
                                                  const int*   __restrict__ season_q,
                                                  const float* __restrict__ year_q,
                                                  const int*   __restrict__ mseason,
                                                  const float* __restrict__ myear,
                                                  float* __restrict__ sim) {
  const int tid = threadIdx.x;
  const int lane = tid & 31, w = tid >> 5;
  const int lm = lane & 15, half = lane >> 4;
  const int m0 = blockIdx.x * 16;
  const int col = m0 + lm;
  const int colc = col < MM ? col : (MM - 1);

  v8f acc = v8f_zero();
  float nsq = 0.f;
  const int kbase = w * (HID / 4);
  const float* ap = qv + (size_t)lm * HID + 2 * half + kbase;
  const float* bp = mem + (size_t)colc * HID + 2 * half + kbase;

  for (int k = 0; k < HID / 4; k += 4) {
    v2f a = *reinterpret_cast<const v2f*>(ap + k);
    v2f bb = *reinterpret_cast<const v2f*>(bp + k);
    nsq += bb.x * bb.x + bb.y * bb.y;
    acc = __builtin_amdgcn_wmma_f32_16x16x4_f32(
        false, a, false, bb, (short)0, acc, false, false);
  }

  __shared__ float cred[4][8][32];
  __shared__ float nred[4][32];
#pragma unroll
  for (int g = 0; g < 8; ++g) cred[w][g][lane] = acc[g];
  nred[w][lane] = nsq;
  __syncthreads();

  if (tid < 32) {
    float cn = 0.f;
#pragma unroll
    for (int ww = 0; ww < 4; ++ww) cn += nred[ww][lm] + nred[ww][lm + 16];
    const int m = m0 + lm;
    if (m < MM) {
      const int   ms = mseason[m];
      const float my = myear[m];
      const float invn = rsqrtf(fmaxf(cn, 1e-24f));
#pragma unroll
      for (int g = 0; g < 8; ++g) {
        float c = 0.f;
#pragma unroll
        for (int ww = 0; ww < 4; ++ww) c += cred[ww][g][lane];
        const int b = g + 8 * half;
        float s = c * invn * rsqrtf(fmaxf(qnorm2[b], 1e-24f));
        const float mask = (season_q[b] == ms) ? 1.f : 0.f;
        s = s * mask + (1.f - mask) * (-10000.f);
        const float dy = fabsf(year_q[b] - my);
        s *= 0.5f + 0.5f * (1.f - expf(-dy * 0.5f));
        sim[b * MPAD + m] = s;
      }
    }
  }
}

// ---------------- kernel 5: top-K=8 per batch row ----------------
__global__ __launch_bounds__(128) void topk_kernel(const float* __restrict__ sim,
                                                   int* __restrict__ topk) {
  const int b = blockIdx.x, t = threadIdx.x;
  __shared__ float sv[MPAD];
  __shared__ float lv[128];
  __shared__ int   li[128];
  for (int m = t; m < MPAD; m += 128) sv[m] = (m < MM) ? sim[b * MPAD + m] : -3e38f;
  __syncthreads();
  for (int r = 0; r < KK; ++r) {
    float bv = -3e38f; int bi = 0;
    for (int m = t; m < MM; m += 128) {
      const float v = sv[m];
      if (v > bv || (v == bv && m < bi)) { bv = v; bi = m; }
    }
    lv[t] = bv; li[t] = bi; __syncthreads();
    for (int s = 64; s > 0; s >>= 1) {
      if (t < s) {
        if (lv[t + s] > lv[t] || (lv[t + s] == lv[t] && li[t + s] < li[t])) {
          lv[t] = lv[t + s]; li[t] = li[t + s];
        }
      }
      __syncthreads();
    }
    if (t == 0) { topk[b * KK + r] = li[0]; sv[li[0]] = -3e38f; }
    __syncthreads();
  }
}

// ---------------- kernel 6: gather + multihead attention + output projections ----------
__global__ __launch_bounds__(128) void attn_kernel(const float* __restrict__ qout,
                                                   const int*   __restrict__ topk,
                                                   const float* __restrict__ mem,
                                                   const float* __restrict__ in_w,
                                                   const float* __restrict__ in_b,
                                                   const float* __restrict__ ow,
                                                   const float* __restrict__ ob,
                                                   const float* __restrict__ pjw,
                                                   const float* __restrict__ pjb,
                                                   float* __restrict__ out) {
  const int n = blockIdx.x, b = blockIdx.y, t = threadIdx.x;
  __shared__ float qrow[DD], kv[KK][DD];
  __shared__ float qh[DD], kh[KK][DD], vh[KK][DD];
  __shared__ float sc[HH][KK];
  __shared__ float ctx[DD], o1[DD];

  const size_t base = ((size_t)b * NN + n) * DD;
  qrow[t] = qout[base + t];

  // async gather of the 8 top-K rows straight into LDS (b128 per lane, 2 passes)
  {
    const int lane32 = t & 31;        // 32 lanes x 16B cover one 128-float row
    const int krow0 = t >> 5;         // 0..3
#pragma unroll
    for (int p = 0; p < 2; ++p) {
      const int k = krow0 + p * 4;
      const int idx = topk[b * KK + k];
      const float* gsrc = mem + ((size_t)idx * NN + n) * DD + lane32 * 4;
      unsigned lds_off = (unsigned)(uintptr_t)(&kv[k][lane32 * 4]);
      asm volatile("global_load_async_to_lds_b128 %0, %1, off"
                   :: "v"(lds_off), "v"(gsrc) : "memory");
    }
    asm volatile("s_wait_asynccnt 0x0" ::: "memory");
  }
  __syncthreads();

  // q/k/v projections (weights L2-resident)
  {
    const float* wq = in_w + (size_t)t * DD;
    const float* wk = in_w + (size_t)(DD + t) * DD;
    const float* wv = in_w + (size_t)(2 * DD + t) * DD;
    float s = 0.f;
    for (int i = 0; i < DD; ++i) s += wq[i] * qrow[i];
    qh[t] = s + in_b[t];
#pragma unroll
    for (int k = 0; k < KK; ++k) {
      float sk = 0.f, sv2 = 0.f;
      for (int i = 0; i < DD; ++i) { sk += wk[i] * kv[k][i]; sv2 += wv[i] * kv[k][i]; }
      kh[k][t] = sk + in_b[DD + t];
      vh[k][t] = sv2 + in_b[2 * DD + t];
    }
  }
  __syncthreads();

  if (t < HH * KK) {
    const int h = t >> 3, k = t & 7;
    float s = 0.f;
    for (int j = 0; j < DD / HH; ++j) s += qh[h * (DD / HH) + j] * kh[k][h * (DD / HH) + j];
    sc[h][k] = s * 0.17677669529663689f;  // 1/sqrt(32)
  }
  __syncthreads();

  if (t < HH) {
    float mx = -3e38f;
    for (int k = 0; k < KK; ++k) mx = fmaxf(mx, sc[t][k]);
    float sum = 0.f;
    for (int k = 0; k < KK; ++k) { const float e = expf(sc[t][k] - mx); sc[t][k] = e; sum += e; }
    const float inv = 1.f / sum;
    for (int k = 0; k < KK; ++k) sc[t][k] *= inv;
  }
  __syncthreads();

  {
    const int h = t >> 5;  // 32 dims per head
    float c = 0.f;
#pragma unroll
    for (int k = 0; k < KK; ++k) c += sc[h][k] * vh[k][t];
    ctx[t] = c;
  }
  __syncthreads();

  {
    const float* r = ow + (size_t)t * DD;
    float s = 0.f;
    for (int i = 0; i < DD; ++i) s += r[i] * ctx[i];
    o1[t] = s + ob[t];
  }
  __syncthreads();

  {
    const float* r = pjw + (size_t)t * DD;
    float s = 0.f;
    for (int i = 0; i < DD; ++i) s += r[i] * o1[i];
    out[base + t] = s + pjb[t];
  }
}

// ---------------- launch ----------------
extern "C" void kernel_launch(void* const* d_in, const int* in_sizes, int n_in,
                              void* d_out, int out_size, void* d_ws, size_t ws_size,
                              hipStream_t stream) {
  const float* x        = (const float*)d_in[0];
  const int*   season_q = (const int*)  d_in[1];
  const float* year_q   = (const float*)d_in[2];
  const float* dw_w     = (const float*)d_in[3];
  const float* dw_b     = (const float*)d_in[4];
  const float* pw_w     = (const float*)d_in[5];
  const float* pw_b     = (const float*)d_in[6];
  const float* ln_g     = (const float*)d_in[7];
  const float* ln_b     = (const float*)d_in[8];
  const float* in_w     = (const float*)d_in[9];
  const float* in_b     = (const float*)d_in[10];
  const float* ow       = (const float*)d_in[11];
  const float* ob       = (const float*)d_in[12];
  const float* pjw      = (const float*)d_in[13];
  const float* pjb      = (const float*)d_in[14];
  const float* mem      = (const float*)d_in[15];
  const int*   mseason  = (const int*)  d_in[16];
  const float* myear    = (const float*)d_in[17];

  char* ws = (char*)d_ws;
  __bf16* actT   = (__bf16*)ws;                      // 16*112*512*2   = 1,835,008
  float*  q_pre  = (float*)(ws + 1835008);           // 16*65536*4    = 4,194,304
  float*  qnorm2 = (float*)(ws + 6029312);           // 16*4
  float*  sim    = (float*)(ws + 6029376);           // 16*1200*4     = 76,800
  int*    topk   = (int*)  (ws + 6106176);           // 16*8*4

  float* out  = (float*)d_out;
  float* qout = out + (size_t)BB * NN * DD;          // second output of the tuple

  init_kernel<<<1, 32, 0, stream>>>(qnorm2);
  conv_kernel<<<dim3(NN, BB), 128, 0, stream>>>(x, dw_w, dw_b, actT);
  gemm_kernel<<<dim3(HID / 16, BB), 32, 0, stream>>>(pw_w, pw_b, actT, q_pre);
  ln_kernel<<<dim3(NN, BB), 128, 0, stream>>>(q_pre, ln_g, ln_b, qout, qnorm2);
  sim_kernel<<<dim3((MM + 15) / 16), 128, 0, stream>>>(qout, mem, qnorm2, season_q, year_q,
                                                       mseason, myear, sim);
  topk_kernel<<<dim3(BB), 128, 0, stream>>>(sim, topk);
  attn_kernel<<<dim3(NN, BB), 128, 0, stream>>>(qout, topk, mem, in_w, in_b, ow, ob,
                                                pjw, pjb, out);
}
